// ConformerBlock_27487790694992
// MI455X (gfx1250) — matrix operand order repaired and checked by reading, not compiled, not run on hardware
//
#include <hip/hip_runtime.h>
#include <hip/hip_bf16.h>

// ---------------------------------------------------------------------------
// Conformer block forward for gfx1250 (MI455X): all GEMMs + attention via
// v_wmma_f32_16x16x32_f16 (f16 inputs, f32 accumulate), global_prefetch_b8
// for next-slab latency hiding.
// ---------------------------------------------------------------------------

#define CB 4
#define CT 2048
#define CD 256
#define CH 8
#define CHD 32
#define CK 31
#define NTOK (CB * CT)   // 8192 tokens

typedef __attribute__((ext_vector_type(16))) _Float16 v16h;
typedef __attribute__((ext_vector_type(8)))  _Float16 v8h;
typedef __attribute__((ext_vector_type(8)))  float    v8f;

// ---------------------------------------------------------------------------
// WMMA fragment helpers (CDNA5 16-bit A 16x32 / B 32x16 VGPR layouts, wave32)
// ---------------------------------------------------------------------------

// A matrix 16x32 (MxK), source is row-major [16][rstride] halves.
__device__ __forceinline__ v16h load_frag_a(const _Float16* p, int rstride) {
  const int lane = threadIdx.x & 31;
  const int hh = lane >> 4, m = lane & 15;
  v16h a;
#pragma unroll
  for (int j = 0; j < 8; ++j) {
    const int k0 = (j << 1) + (hh << 4);
    a[2 * j]     = p[m * rstride + k0];
    a[2 * j + 1] = p[m * rstride + k0 + 1];
  }
  return a;
}

// B matrix 32x16 (KxN), source stored TRANSPOSED as [n][rstride] halves (k fast).
__device__ __forceinline__ v16h load_frag_b_t(const _Float16* p, int rstride) {
  const int lane = threadIdx.x & 31;
  const int hh = lane >> 4, n = lane & 15;
  v16h b;
#pragma unroll
  for (int j = 0; j < 8; ++j) {
    const int k0 = (j << 1) + (hh << 4);
    b[2 * j]     = p[n * rstride + k0];
    b[2 * j + 1] = p[n * rstride + k0 + 1];
  }
  return b;
}

__device__ __forceinline__ float silu_f(float x) {
  return x / (1.0f + __expf(-x));
}

// ---------------------------------------------------------------------------
// Generic tiled GEMM:  out = act(A[N,K] @ W[K,Dout] + bias)
// MODE 0: f16 out (bias only)          -> QKV
// MODE 1: f16 out (bias + SiLU)        -> FF first matmul
// MODE 2: f32 out = res + scale*(..)   -> residual matmuls / raw pw1
// Grid: (Dout/128, N/64), block 128 (4 waves). Block tile 64 rows x 128 cols;
// each wave owns a 32x64 quadrant => 8 WMMAs per 32-wide K step.
// ---------------------------------------------------------------------------

constexpr int LDA = 40;  // padded LDS row stride (halves)
constexpr int LDB = 40;

template <int MODE>
__global__ void __launch_bounds__(128)
gemm_kernel(const _Float16* __restrict__ A, const _Float16* __restrict__ W,
            const float* __restrict__ bias, const float* __restrict__ res,
            float scale, _Float16* __restrict__ out16, float* __restrict__ out32,
            int Kdim, int Dout) {
  __shared__ __align__(16) _Float16 As[64 * LDA];
  __shared__ __align__(16) _Float16 Bs[128 * LDB];

  const int tid     = threadIdx.x;
  const int colBase = blockIdx.x * 128;
  const int rowBase = blockIdx.y * 64;
  const int wave    = tid >> 5;
  const int wr      = (wave >> 1) << 5;   // wave row offset in tile (0/32)
  const int wc      = (wave & 1) << 6;    // wave col offset in tile (0/64)
  const int lane    = tid & 31;
  const int l16     = lane & 15;
  const int hh      = lane >> 4;

  v8f acc[2][4] = {{{}, {}, {}, {}}, {{}, {}, {}, {}}};

  const int ar = tid >> 1;            // A-tile row this thread loads (0..63)
  const int ac = (tid & 1) << 4;      // 16-half chunk within the 32-wide K slab

  for (int k0 = 0; k0 < Kdim; k0 += 32) {
    // ---- prefetch next K slab while this one is staged/consumed ----
    if (k0 + 32 < Kdim) {
      __builtin_prefetch(A + (size_t)(rowBase + ar) * Kdim + (k0 + 32) + ac, 0, 3);
      __builtin_prefetch(W + (size_t)(k0 + 32 + (tid >> 2)) * Dout + colBase + ((tid & 3) << 5), 0, 3);
    }
    // ---- stage A tile (64 rows x 32 k) ----
    {
      const v8h* src = (const v8h*)(A + (size_t)(rowBase + ar) * Kdim + k0 + ac);
      v8h x0 = src[0];
      v8h x1 = src[1];
      *(v8h*)&As[ar * LDA + ac]     = x0;
      *(v8h*)&As[ar * LDA + ac + 8] = x1;
    }
    // ---- stage W tile transposed: Bs[n][k] (128 cols x 32 k) ----
#pragma unroll
    for (int it = 0; it < 4; ++it) {
      const int slot = tid + (it << 7);      // 0..511
      const int kk   = slot >> 4;            // 0..31
      const int cseg = (slot & 15) << 3;     // 0,8,..,120
      const v8h* src = (const v8h*)(W + (size_t)(k0 + kk) * Dout + colBase + cseg);
      v8h x = src[0];
#pragma unroll
      for (int j = 0; j < 8; ++j) Bs[(cseg + j) * LDB + kk] = x[j];
    }
    __syncthreads();

    v16h a0 = load_frag_a(&As[(wr + 0)  * LDA], LDA);
    v16h a1 = load_frag_a(&As[(wr + 16) * LDA], LDA);
    v16h b0 = load_frag_b_t(&Bs[(wc + 0)  * LDB], LDB);
    v16h b1 = load_frag_b_t(&Bs[(wc + 16) * LDB], LDB);
    v16h b2 = load_frag_b_t(&Bs[(wc + 32) * LDB], LDB);
    v16h b3 = load_frag_b_t(&Bs[(wc + 48) * LDB], LDB);

    acc[0][0] = __builtin_amdgcn_wmma_f32_16x16x32_f16(false, a0, false, b0, (short)0, acc[0][0], false, false);
    acc[0][1] = __builtin_amdgcn_wmma_f32_16x16x32_f16(false, a0, false, b1, (short)0, acc[0][1], false, false);
    acc[0][2] = __builtin_amdgcn_wmma_f32_16x16x32_f16(false, a0, false, b2, (short)0, acc[0][2], false, false);
    acc[0][3] = __builtin_amdgcn_wmma_f32_16x16x32_f16(false, a0, false, b3, (short)0, acc[0][3], false, false);
    acc[1][0] = __builtin_amdgcn_wmma_f32_16x16x32_f16(false, a1, false, b0, (short)0, acc[1][0], false, false);
    acc[1][1] = __builtin_amdgcn_wmma_f32_16x16x32_f16(false, a1, false, b1, (short)0, acc[1][1], false, false);
    acc[1][2] = __builtin_amdgcn_wmma_f32_16x16x32_f16(false, a1, false, b2, (short)0, acc[1][2], false, false);
    acc[1][3] = __builtin_amdgcn_wmma_f32_16x16x32_f16(false, a1, false, b3, (short)0, acc[1][3], false, false);
    __syncthreads();
  }

  // ---- epilogue ----
#pragma unroll
  for (int mi = 0; mi < 2; ++mi) {
#pragma unroll
    for (int ni = 0; ni < 4; ++ni) {
      const int gc = colBase + wc + (ni << 4) + l16;
      const float bv = bias[gc];
#pragma unroll
      for (int v = 0; v < 8; ++v) {
        const int row = rowBase + wr + (mi << 4) + (hh << 3) + v;
        const float val = acc[mi][ni][v] + bv;
        const size_t idx = (size_t)row * Dout + gc;
        if (MODE == 0) {
          out16[idx] = (_Float16)val;
        } else if (MODE == 1) {
          out16[idx] = (_Float16)silu_f(val);
        } else {
          float o = scale * val;
          if (res) o += res[idx];
          out32[idx] = o;
        }
      }
    }
  }
}

// ---------------------------------------------------------------------------
// Flash-style MHSA: one wave per (b, h, 16-query tile). HD=32 so QK^T is a
// single K-step WMMA; PV accumulates over 32-key blocks with online softmax.
// qkv layout: [NTOK, 768] f16, cols = qkv_sel*256 + h*32 + d.
// ---------------------------------------------------------------------------

__global__ void __launch_bounds__(32)
attn_kernel(const _Float16* __restrict__ qkv, _Float16* __restrict__ attnout) {
  const int bh = blockIdx.x;
  const int b  = bh / CH;
  const int h  = bh % CH;
  const int q0 = blockIdx.y * 16;

  const int lane = threadIdx.x & 31;
  const int l16  = lane & 15;
  const int hh   = lane >> 4;

  __shared__ __align__(16) _Float16 Ptile[16 * 32];

  // Q fragment (A layout), with 1/sqrt(HD) folded in.
  const float qscale = 0.17677669529663687f;  // 1/sqrt(32)
  const _Float16* qbase = qkv + (size_t)(b * CT + q0) * 768 + h * 32;
  v16h qa;
#pragma unroll
  for (int j = 0; j < 8; ++j) {
    const int k0 = (j << 1) + (hh << 4);
    qa[2 * j]     = (_Float16)((float)qbase[l16 * 768 + k0]     * qscale);
    qa[2 * j + 1] = (_Float16)((float)qbase[l16 * 768 + k0 + 1] * qscale);
  }

  float mrow[8], lrow[8];
#pragma unroll
  for (int v = 0; v < 8; ++v) { mrow[v] = -1.0e30f; lrow[v] = 0.0f; }
  v8f o0 = {}, o1 = {};
  const v8f zf = {};

  for (int kb = 0; kb < CT; kb += 32) {
    // ---- K block: two B fragments (keys kb..kb+15, kb+16..kb+31) ----
    const _Float16* kbase = qkv + (size_t)(b * CT + kb) * 768 + 256 + h * 32;
    const _Float16* vbase = qkv + (size_t)(b * CT + kb) * 768 + 512 + h * 32;
    if (kb + 32 < CT) {  // prefetch next key/value block (global_prefetch_b8)
      __builtin_prefetch(kbase + (size_t)32 * 768 + lane * 768, 0, 3);
      __builtin_prefetch(vbase + (size_t)32 * 768 + lane * 768, 0, 3);
    }
    v16h kf0, kf1;
#pragma unroll
    for (int j = 0; j < 8; ++j) {
      const int d0 = (j << 1) + (hh << 4);
      kf0[2 * j]     = kbase[l16 * 768 + d0];
      kf0[2 * j + 1] = kbase[l16 * 768 + d0 + 1];
      kf1[2 * j]     = kbase[(l16 + 16) * 768 + d0];
      kf1[2 * j + 1] = kbase[(l16 + 16) * 768 + d0 + 1];
    }
    v8f s0 = __builtin_amdgcn_wmma_f32_16x16x32_f16(false, qa, false, kf0, (short)0, zf, false, false);
    v8f s1 = __builtin_amdgcn_wmma_f32_16x16x32_f16(false, qa, false, kf1, (short)0, zf, false, false);

    // ---- online softmax over this 32-key slab ----
    float p0[8], p1[8];
#pragma unroll
    for (int v = 0; v < 8; ++v) {
      float tmx = fmaxf(s0[v], s1[v]);
#pragma unroll
      for (int off = 1; off < 16; off <<= 1) tmx = fmaxf(tmx, __shfl_xor(tmx, off, 32));
      const float newm = fmaxf(mrow[v], tmx);
      const float corr = __expf(mrow[v] - newm);
      p0[v] = __expf(s0[v] - newm);
      p1[v] = __expf(s1[v] - newm);
      float rs = p0[v] + p1[v];
#pragma unroll
      for (int off = 1; off < 16; off <<= 1) rs += __shfl_xor(rs, off, 32);
      lrow[v] = lrow[v] * corr + rs;
      mrow[v] = newm;
      o0[v] *= corr;
      o1[v] *= corr;
    }

    // ---- transpose P (C layout -> A layout) through LDS ----
#pragma unroll
    for (int v = 0; v < 8; ++v) {
      Ptile[(v + (hh << 3)) * 32 + l16]      = (_Float16)p0[v];
      Ptile[(v + (hh << 3)) * 32 + 16 + l16] = (_Float16)p1[v];
    }
    __syncthreads();
    v16h pa;
#pragma unroll
    for (int j = 0; j < 8; ++j) {
      const int k0 = (j << 1) + (hh << 4);
      pa[2 * j]     = Ptile[l16 * 32 + k0];
      pa[2 * j + 1] = Ptile[l16 * 32 + k0 + 1];
    }
    __syncthreads();

    // ---- V block: B fragments (K-dim = key index, N = head dim) ----
    v16h vf0, vf1;
#pragma unroll
    for (int j = 0; j < 8; ++j) {
      const int kk = (j << 1) + (hh << 4);
      vf0[2 * j]     = vbase[kk * 768 + l16];
      vf0[2 * j + 1] = vbase[(kk + 1) * 768 + l16];
      vf1[2 * j]     = vbase[kk * 768 + 16 + l16];
      vf1[2 * j + 1] = vbase[(kk + 1) * 768 + 16 + l16];
    }
    o0 = __builtin_amdgcn_wmma_f32_16x16x32_f16(false, pa, false, vf0, (short)0, o0, false, false);
    o1 = __builtin_amdgcn_wmma_f32_16x16x32_f16(false, pa, false, vf1, (short)0, o1, false, false);
  }

  // ---- finalize: divide by l and scatter to [NTOK, 256] f16 ----
  _Float16* obase = attnout + (size_t)(b * CT + q0) * 256 + h * 32;
#pragma unroll
  for (int v = 0; v < 8; ++v) {
    const float inv = 1.0f / lrow[v];
    obase[(v + (hh << 3)) * 256 + l16]      = (_Float16)(o0[v] * inv);
    obase[(v + (hh << 3)) * 256 + 16 + l16] = (_Float16)(o1[v] * inv);
  }
}

// ---------------------------------------------------------------------------
// Elementwise / reduction glue kernels
// ---------------------------------------------------------------------------

__global__ void cvt_kernel(const float* __restrict__ s, _Float16* __restrict__ d, int n) {
  const int i = blockIdx.x * 256 + threadIdx.x;
  if (i < n) d[i] = (_Float16)s[i];
}

template <bool OUT16>
__global__ void __launch_bounds__(256)
ln_kernel(const float* __restrict__ x, const float* __restrict__ g,
          const float* __restrict__ bb, _Float16* __restrict__ o16,
          float* __restrict__ o32) {
  const int n = blockIdx.x;
  const int c = threadIdx.x;
  __shared__ float red[256];
  const float v = x[(size_t)n * CD + c];
  red[c] = v;
  __syncthreads();
  for (int s = 128; s > 0; s >>= 1) {
    if (c < s) red[c] += red[c + s];
    __syncthreads();
  }
  const float mu = red[0] * (1.0f / CD);
  __syncthreads();
  const float d = v - mu;
  red[c] = d * d;
  __syncthreads();
  for (int s = 128; s > 0; s >>= 1) {
    if (c < s) red[c] += red[c + s];
    __syncthreads();
  }
  const float var = red[0] * (1.0f / CD);
  const float y = d * rsqrtf(var + 1e-5f) * g[c] + bb[c];
  if (OUT16) o16[(size_t)n * CD + c] = (_Float16)y;
  else       o32[(size_t)n * CD + c] = y;
}

__global__ void glu_kernel(const float* __restrict__ raw, _Float16* __restrict__ o) {
  const int n = blockIdx.x;
  const int c = threadIdx.x;
  const float a = raw[(size_t)n * 512 + c];
  const float gt = raw[(size_t)n * 512 + 256 + c];
  o[(size_t)n * CD + c] = (_Float16)(a / (1.0f + __expf(-gt)));
}

__global__ void dwconv_kernel(const _Float16* __restrict__ glu,
                              const float* __restrict__ w,
                              const float* __restrict__ bias,
                              float* __restrict__ out) {
  const int n = blockIdx.x;
  const int c = threadIdx.x;
  const int b = n / CT;
  const int t = n % CT;
  float acc = bias[c];
#pragma unroll
  for (int k = 0; k < CK; ++k) {
    const int tt = t + k - CK / 2;
    if (tt >= 0 && tt < CT)
      acc += (float)glu[(size_t)(b * CT + tt) * CD + c] * w[c * CK + k];
  }
  out[(size_t)n * CD + c] = acc;
}

__global__ void zero_kernel(float* p) { p[threadIdx.x] = 0.0f; }

__global__ void bnstats_kernel(const float* __restrict__ h, float* __restrict__ stat) {
  const int c = threadIdx.x;
  const int blk = blockIdx.x;       // 256 blocks x 32 tokens
  float s = 0.0f, s2 = 0.0f;
  for (int i = 0; i < 32; ++i) {
    const float v = h[(size_t)(blk * 32 + i) * CD + c];
    s += v;
    s2 += v * v;
  }
  atomicAdd(&stat[c], s);
  atomicAdd(&stat[CD + c], s2);
}

__global__ void bnapply_kernel(const float* __restrict__ h, const float* __restrict__ stat,
                               const float* __restrict__ g, const float* __restrict__ bb,
                               _Float16* __restrict__ o) {
  const int n = blockIdx.x;
  const int c = threadIdx.x;
  const float mean = stat[c] * (1.0f / NTOK);
  const float var = stat[CD + c] * (1.0f / NTOK) - mean * mean;
  const float y = (h[(size_t)n * CD + c] - mean) * rsqrtf(var + 1e-5f) * g[c] + bb[c];
  o[(size_t)n * CD + c] = (_Float16)silu_f(y);
}

// ---------------------------------------------------------------------------
// Host-side orchestration
// ---------------------------------------------------------------------------

extern "C" void kernel_launch(void* const* d_in, const int* in_sizes, int n_in,
                              void* d_out, int out_size, void* d_ws, size_t ws_size,
                              hipStream_t stream) {
  const float* x         = (const float*)d_in[0];
  const float* ff1_ln_g  = (const float*)d_in[1];
  const float* ff1_ln_b  = (const float*)d_in[2];
  const float* ff1_w1    = (const float*)d_in[3];
  const float* ff1_b1    = (const float*)d_in[4];
  const float* ff1_w2    = (const float*)d_in[5];
  const float* ff1_b2    = (const float*)d_in[6];
  const float* mhsa_ln_g = (const float*)d_in[7];
  const float* mhsa_ln_b = (const float*)d_in[8];
  const float* qkv_w     = (const float*)d_in[9];
  const float* qkv_b     = (const float*)d_in[10];
  const float* out_w     = (const float*)d_in[11];
  const float* out_b     = (const float*)d_in[12];
  const float* conv_ln_g = (const float*)d_in[13];
  const float* conv_ln_b = (const float*)d_in[14];
  const float* pw1_w     = (const float*)d_in[15];
  const float* pw1_b     = (const float*)d_in[16];
  const float* dw_w      = (const float*)d_in[17];
  const float* dw_b      = (const float*)d_in[18];
  const float* bn_g      = (const float*)d_in[19];
  const float* bn_b      = (const float*)d_in[20];
  const float* pw2_w     = (const float*)d_in[21];
  const float* pw2_b     = (const float*)d_in[22];
  const float* ff2_ln_g  = (const float*)d_in[23];
  const float* ff2_ln_b  = (const float*)d_in[24];
  const float* ff2_w1    = (const float*)d_in[25];
  const float* ff2_b1    = (const float*)d_in[26];
  const float* ff2_w2    = (const float*)d_in[27];
  const float* ff2_b2    = (const float*)d_in[28];
  const float* fin_ln_g  = (const float*)d_in[29];
  const float* fin_ln_b  = (const float*)d_in[30];
  (void)in_sizes; (void)n_in; (void)out_size; (void)ws_size;

  char* ws = (char*)d_ws;
  size_t off = 0;
  auto alloc = [&](size_t bytes) -> char* {
    off = (off + 255) & ~(size_t)255;
    char* p = ws + off;
    off += bytes;
    return p;
  };

  // f16 weights
  const float* wsrc[8] = {ff1_w1, ff1_w2, qkv_w, out_w, pw1_w, pw2_w, ff2_w1, ff2_w2};
  const int wn[8] = {CD * 512, 512 * CD, CD * 768, CD * CD, CD * 512, CD * CD, CD * 512, 512 * CD};
  _Float16* wf[8];
  for (int i = 0; i < 8; ++i) wf[i] = (_Float16*)alloc((size_t)wn[i] * 2);

  float*     X0     = (float*)alloc((size_t)NTOK * CD * 4);
  float*     X1     = (float*)alloc((size_t)NTOK * CD * 4);
  _Float16*  LN16   = (_Float16*)alloc((size_t)NTOK * CD * 2);
  _Float16*  BIG16  = (_Float16*)alloc((size_t)NTOK * 768 * 2);
  _Float16*  SM16   = (_Float16*)alloc((size_t)NTOK * CD * 2);
  float*     F32BIG = (float*)alloc((size_t)NTOK * 512 * 4);
  float*     BNST   = (float*)alloc(512 * 4);

  // 0) weights -> f16
  for (int i = 0; i < 8; ++i)
    cvt_kernel<<<(wn[i] + 255) / 256, 256, 0, stream>>>(wsrc[i], wf[i], wn[i]);

  // 1) FF1
  ln_kernel<true><<<NTOK, 256, 0, stream>>>(x, ff1_ln_g, ff1_ln_b, LN16, nullptr);
  gemm_kernel<1><<<dim3(4, 128), 128, 0, stream>>>(LN16, wf[0], ff1_b1, nullptr, 0.f, BIG16, nullptr, 256, 512);
  gemm_kernel<2><<<dim3(2, 128), 128, 0, stream>>>(BIG16, wf[1], ff1_b2, x, 0.5f, nullptr, X0, 512, 256);

  // 2) MHSA
  ln_kernel<true><<<NTOK, 256, 0, stream>>>(X0, mhsa_ln_g, mhsa_ln_b, LN16, nullptr);
  gemm_kernel<0><<<dim3(6, 128), 128, 0, stream>>>(LN16, wf[2], qkv_b, nullptr, 0.f, BIG16, nullptr, 256, 768);
  attn_kernel<<<dim3(CB * CH, CT / 16), 32, 0, stream>>>(BIG16, SM16);
  gemm_kernel<2><<<dim3(2, 128), 128, 0, stream>>>(SM16, wf[3], out_b, X0, 1.0f, nullptr, X1, 256, 256);

  // 3) Conv module
  ln_kernel<true><<<NTOK, 256, 0, stream>>>(X1, conv_ln_g, conv_ln_b, LN16, nullptr);
  gemm_kernel<2><<<dim3(4, 128), 128, 0, stream>>>(LN16, wf[4], pw1_b, nullptr, 1.0f, nullptr, F32BIG, 256, 512);
  glu_kernel<<<NTOK, 256, 0, stream>>>(F32BIG, SM16);
  dwconv_kernel<<<NTOK, 256, 0, stream>>>(SM16, dw_w, dw_b, F32BIG);  // reuse as [NTOK,256] f32
  zero_kernel<<<1, 512, 0, stream>>>(BNST);
  bnstats_kernel<<<256, 256, 0, stream>>>(F32BIG, BNST);
  bnapply_kernel<<<NTOK, 256, 0, stream>>>(F32BIG, BNST, bn_g, bn_b, LN16);
  gemm_kernel<2><<<dim3(2, 128), 128, 0, stream>>>(LN16, wf[5], pw2_b, X1, 1.0f, nullptr, X0, 256, 256);

  // 4) FF2
  ln_kernel<true><<<NTOK, 256, 0, stream>>>(X0, ff2_ln_g, ff2_ln_b, LN16, nullptr);
  gemm_kernel<1><<<dim3(4, 128), 128, 0, stream>>>(LN16, wf[6], ff2_b1, nullptr, 0.f, BIG16, nullptr, 256, 512);
  gemm_kernel<2><<<dim3(2, 128), 128, 0, stream>>>(BIG16, wf[7], ff2_b2, X0, 0.5f, nullptr, X1, 512, 256);

  // 5) Final LN -> f32 output
  ln_kernel<false><<<NTOK, 256, 0, stream>>>(X1, fin_ln_g, fin_ln_b, nullptr, (float*)d_out);
}
